// SelfAttention_6107443495076
// MI455X (gfx1250) — matrix operand order repaired and checked by reading, not compiled, not run on hardware
//
#include <hip/hip_runtime.h>

// Problem dims (from reference):
#define BS   2
#define SEQ  2048
#define DIN  1024
#define NH   16
#define DH   64
#define NOUT (NH * DH)   // 1024

typedef __attribute__((ext_vector_type(16))) _Float16 v16h;
typedef __attribute__((ext_vector_type(8)))  _Float16 v8h;
typedef __attribute__((ext_vector_type(8)))  float    v8f;
typedef __attribute__((ext_vector_type(4)))  unsigned v4u;
typedef __attribute__((ext_vector_type(8)))  int      v8i;
typedef __attribute__((ext_vector_type(4)))  int      v4i;

static __device__ __forceinline__ v16h cat16(v8h lo, v8h hi) {
  return __builtin_shufflevector(lo, hi, 0,1,2,3,4,5,6,7,8,9,10,11,12,13,14,15);
}

static __device__ __forceinline__ v8f wmma_f16(v16h a, v16h b, v8f c) {
  // 8 args: (neg_a, A, neg_b, B, c_mod, C, reuse_a, reuse_b)
  return __builtin_amdgcn_wmma_f32_16x16x32_f16(false, a, false, b, (short)0, c, false, false);
}

// ---------------------------------------------------------------------------
// Tensor Data Mover: 2-D f16 tile (tile_d0 contiguous elems per row,
// tile_d1 rows, row stride stride0 elems) from global -> LDS.
// D# packed per CDNA5 ISA 8.3/8.4: group0 = {flags/count, lds_addr,
// global_addr lo, global_addr hi | type=2}; group1 = {data_size=2B,
// tensor dims == tile dims (tile fully in-bounds), strides}; groups 2/3
// (and the trailing group on this toolchain's 6-arg builtin) zeroed.
// ---------------------------------------------------------------------------
static __device__ __forceinline__ void tdm_load_2d_f16(
    unsigned lds_off, unsigned long long gaddr,
    unsigned tile_d0, unsigned tile_d1, unsigned long long stride0_elems)
{
  v4u g0;
  g0[0] = 1u;                                        // count=1, user mode
  g0[1] = lds_off;                                   // LDS byte address
  g0[2] = (unsigned)gaddr;                           // global addr [31:0]
  g0[3] = (unsigned)((gaddr >> 32) & 0x01FFFFFFull)  // global addr [56:32]
          | (2u << 30);                              // type = 2 ("image")
  v8i g1;
  g1[0] = (int)(1u << 16);                           // wg_mask=0, data_size=2B
  g1[1] = (int)((tile_d0 & 0xFFFFu) << 16);          // tensor_dim0 lo16
  g1[2] = (int)(((tile_d0 >> 16) & 0xFFFFu)          // tensor_dim0 hi16
          | ((tile_d1 & 0xFFFFu) << 16));            // tensor_dim1 lo16
  g1[3] = (int)(((tile_d1 >> 16) & 0xFFFFu)          // tensor_dim1 hi16
          | ((tile_d0 & 0xFFFFu) << 16));            // tile_dim0
  g1[4] = (int)(tile_d1 & 0xFFFFu);                  // tile_dim1; tile_dim2=0
  g1[5] = (int)(stride0_elems & 0xFFFFFFFFull);      // tensor_dim0_stride lo
  g1[6] = (int)((stride0_elems >> 32) & 0xFFFFull);  // stride0 hi; stride1=0
  g1[7] = 0;
  v4i gz4 = {0, 0, 0, 0};
  v8i gz8 = {0, 0, 0, 0, 0, 0, 0, 0};
  // 6-arg form on this toolchain: (g0, g1, g2, g3, extra, cpol)
  __builtin_amdgcn_tensor_load_to_lds(g0, g1, gz4, gz4, gz8, 0);
}

static __device__ __forceinline__ unsigned lds_offset_of(const void* p) {
  // generic pointer to LDS: addr[31:0] is the LDS offset (aperture in hi bits)
  return (unsigned)(unsigned long long)p;
}

// ---------------------------------------------------------------------------
// Kernel 0: transpose + convert weights.  W [k][n] f32 -> Wt [n][k] f16,
// via a 32x32 LDS tile so both global reads and writes are coalesced.
// ---------------------------------------------------------------------------
__global__ __launch_bounds__(256)
void wt_transpose_kernel(const float* __restrict__ Wq,
                         const float* __restrict__ Wk,
                         const float* __restrict__ Wv,
                         _Float16* __restrict__ wt)
{
  __shared__ float tile[32][33];
  const int mat = blockIdx.z;
  const float* W = (mat == 0) ? Wq : (mat == 1) ? Wk : Wv;
  const int k0 = blockIdx.x * 32;
  const int n0 = blockIdx.y * 32;
  const int tx = threadIdx.x;       // 0..31
  const int ty = threadIdx.y;       // 0..7

  #pragma unroll
  for (int r = ty; r < 32; r += 8)
    tile[r][tx] = W[(size_t)(k0 + r) * NOUT + n0 + tx];
  __syncthreads();

  _Float16* dst = wt + ((size_t)mat << 20);   // 1024*1024 per matrix
  #pragma unroll
  for (int r = ty; r < 32; r += 8)
    dst[(size_t)(n0 + r) * DIN + k0 + tx] = (_Float16)tile[tx][r];
}

// ---------------------------------------------------------------------------
// Kernel 1: QKV projection GEMM.  One wave computes a 16(M) x 64(N) strip of
// (x @ W + b): A fragment (x, f32->f16) loaded once per K-step and reused by
// 4 WMMAs; B fragments are two 16B loads from the pre-transposed f16 Wt.
//
// A layout (16-bit A 16x32): lanes 0-15: M=lane, K in {0..7,16..23};
//                            lanes16-31: M=lane-16, K in {8..15,24..31}.
// B layout (16-bit B 32x16): lanes 0-15: N=lane, K in {0..7,16..23};
//                            lanes16-31: N=lane-16, K in {8..15,24..31}.
// Q/K stored f16 [bh][s][d]; V stored transposed f16 [bh][d][s].
// ---------------------------------------------------------------------------
__global__ __launch_bounds__(32)
void qkv_proj_kernel(const float* __restrict__ x,
                     const _Float16* __restrict__ wt,
                     const float* __restrict__ bq,
                     const float* __restrict__ bk,
                     const float* __restrict__ bv,
                     _Float16* __restrict__ qw, _Float16* __restrict__ kw,
                     _Float16* __restrict__ vt)
{
  const int lane = threadIdx.x;
  const int hb   = lane >> 4;
  const int lr   = lane & 15;
  const int row0 = blockIdx.x * 16;   // M tile over B*SEQ
  const int col0 = blockIdx.y * 64;   // N strip over H*DH
  const int mat  = blockIdx.z;        // 0=Q 1=K 2=V

  const float* bias = (mat == 0) ? bq : (mat == 1) ? bk : bv;
  const _Float16* wtm = wt + ((size_t)mat << 20);
  const float* xrow = x + (size_t)(row0 + lr) * DIN;

  // per-lane row pointers for the 4 B-columns of this strip
  const _Float16* wrow[4];
  #pragma unroll
  for (int t = 0; t < 4; ++t)
    wrow[t] = wtm + (size_t)(col0 + t * 16 + lr) * DIN;

  v8f acc[4] = {v8f{}, v8f{}, v8f{}, v8f{}};
  for (int kk = 0; kk < DIN; kk += 32) {
    // ---- A fragment: 4 contiguous float4 loads, packed converts
    const float4 a0 = *(const float4*)(xrow + kk + hb * 8);
    const float4 a1 = *(const float4*)(xrow + kk + hb * 8 + 4);
    const float4 a2 = *(const float4*)(xrow + kk + 16 + hb * 8);
    const float4 a3 = *(const float4*)(xrow + kk + 16 + hb * 8 + 4);
    v16h a;
    a[0]  = (_Float16)a0.x; a[1]  = (_Float16)a0.y;
    a[2]  = (_Float16)a0.z; a[3]  = (_Float16)a0.w;
    a[4]  = (_Float16)a1.x; a[5]  = (_Float16)a1.y;
    a[6]  = (_Float16)a1.z; a[7]  = (_Float16)a1.w;
    a[8]  = (_Float16)a2.x; a[9]  = (_Float16)a2.y;
    a[10] = (_Float16)a2.z; a[11] = (_Float16)a2.w;
    a[12] = (_Float16)a3.x; a[13] = (_Float16)a3.y;
    a[14] = (_Float16)a3.z; a[15] = (_Float16)a3.w;

    if (kk + 32 < DIN)
      __builtin_prefetch(wrow[0] + kk + 32, 0, 0);

    // ---- 4 WMMAs sharing the A fragment; B = two 16B loads each
    #pragma unroll
    for (int t = 0; t < 4; ++t) {
      const v16h b = cat16(*(const v8h*)(wrow[t] + kk + hb * 8), *(const v8h*)(wrow[t] + kk + 16 + hb * 8));
      acc[t] = wmma_f16(a, b, acc[t]);
    }
  }

  // ---- epilogue: bias, Q scale (1/sqrt(DH)), store f16
  #pragma unroll
  for (int t = 0; t < 4; ++t) {
    const int n  = col0 + t * 16 + lr;
    const float bn = bias[n];
    const int hh = n >> 6;
    const int dd = n & 63;
    #pragma unroll
    for (int r = 0; r < 8; ++r) {
      const int m  = row0 + hb * 8 + r;
      const int bb = m >> 11;            // / SEQ
      const int ss = m & (SEQ - 1);
      float v = acc[t][r] + bn;
      if (mat == 0) v *= 0.125f;         // 1/sqrt(64)
      const _Float16 hv = (_Float16)v;
      const int bh = bb * NH + hh;
      if (mat == 0)      qw[((size_t)bh * SEQ + ss) * DH + dd] = hv;
      else if (mat == 1) kw[((size_t)bh * SEQ + ss) * DH + dd] = hv;
      else               vt[((size_t)bh * DH + dd) * SEQ + ss] = hv;
    }
  }
}

// ---------------------------------------------------------------------------
// Kernel 2: flash attention with TDM double buffering.  One wave per
// (b,h) x 16-query tile, keys streamed in blocks of 32:
//   TDM loads K-tile [32 keys][64 d] and V^T-tile [64 d][32 keys] into LDS
//   (double buffered, s_wait_tensorcnt pipelining), then 4 WMMAs for
//   S = Q K^T, online softmax with cross-lane reductions, P transposed
//   through LDS, 4 WMMAs for O += P V.
// ---------------------------------------------------------------------------
__global__ __launch_bounds__(32)
void flash_attn_kernel(const float* __restrict__ mask,
                       const _Float16* __restrict__ qw,
                       const _Float16* __restrict__ kw,
                       const _Float16* __restrict__ vt,
                       float* __restrict__ out)
{
  __shared__ __align__(64) _Float16 kbuf[2][32 * DH];   // 4KB each
  __shared__ __align__(64) _Float16 vbuf[2][DH * 32];   // 4KB each
  __shared__ __align__(64) _Float16 plds[16 * 32];      // 1KB

  const int lane = threadIdx.x;
  const int hb   = lane >> 4;
  const int lr   = lane & 15;
  const int bh   = blockIdx.y;       // b*NH + h
  const int b    = bh >> 4;
  const int h    = bh & 15;
  const int q0   = blockIdx.x * 16;

  // Q fragments for d in [0,32) and [32,64); loaded once.
  const _Float16* qrow = qw + ((size_t)bh * SEQ + q0 + lr) * DH;
  const v16h aq0 = cat16(*(const v8h*)(qrow + hb * 16),
                         *(const v8h*)(qrow + hb * 16 + 8));
  const v16h aq1 = cat16(*(const v8h*)(qrow + 32 + hb * 16),
                         *(const v8h*)(qrow + 40 + hb * 16));

  const _Float16* kbase = kw + (size_t)bh * SEQ * DH;
  const _Float16* vbase = vt + (size_t)bh * DH * SEQ;
  const float*    mrow  = mask + (size_t)b * SEQ;

  const unsigned koff[2] = { lds_offset_of(&kbuf[0][0]), lds_offset_of(&kbuf[1][0]) };
  const unsigned voff[2] = { lds_offset_of(&vbuf[0][0]), lds_offset_of(&vbuf[1][0]) };

  float mrun[8], lrun[8];
  v8f o[4] = {v8f{}, v8f{}, v8f{}, v8f{}};
  #pragma unroll
  for (int r = 0; r < 8; ++r) { mrun[r] = -1e30f; lrun[r] = 0.0f; }

  // stage key/value block 0
  tdm_load_2d_f16(koff[0], (unsigned long long)kbase, DH, 32, DH);
  tdm_load_2d_f16(voff[0], (unsigned long long)vbase, 32, DH, SEQ);

  for (int s0 = 0; s0 < SEQ; s0 += 32) {
    const int cur = (s0 >> 5) & 1;
    if (s0 + 32 < SEQ) {
      const int nxt = cur ^ 1;
      tdm_load_2d_f16(koff[nxt],
                      (unsigned long long)(kbase + (size_t)(s0 + 32) * DH),
                      DH, 32, DH);
      tdm_load_2d_f16(voff[nxt],
                      (unsigned long long)(vbase + s0 + 32),
                      32, DH, SEQ);
      // TDM completes in order: <=2 outstanding => current block resident
      __builtin_amdgcn_s_wait_tensorcnt(2);
    } else {
      __builtin_amdgcn_s_wait_tensorcnt(0);
    }

    // ---- scores: two 16x16 tiles from the LDS K tile
    const _Float16* k0 = &kbuf[cur][(0 + lr) * DH + hb * 16];
    const _Float16* k1 = &kbuf[cur][(16 + lr) * DH + hb * 16];
    v8f st0 = {}, st1 = {};
    st0 = wmma_f16(aq0, *(const v16h*)(k0),      st0);
    st0 = wmma_f16(aq1, *(const v16h*)(k0 + 32), st0);
    st1 = wmma_f16(aq0, *(const v16h*)(k1),      st1);
    st1 = wmma_f16(aq1, *(const v16h*)(k1 + 32), st1);

    const float mv0 = mrow[s0 + lr];
    const float mv1 = mrow[s0 + 16 + lr];

    // ---- online softmax: lane owns rows m = hb*8 + r at column n = lr
    float alpha[8];
    #pragma unroll
    for (int r = 0; r < 8; ++r) {
      float x0 = st0[r] + mv0;
      float x1 = st1[r] + mv1;
      float mx = fmaxf(x0, x1);
      mx = fmaxf(mx, __shfl_xor(mx, 1));
      mx = fmaxf(mx, __shfl_xor(mx, 2));
      mx = fmaxf(mx, __shfl_xor(mx, 4));
      mx = fmaxf(mx, __shfl_xor(mx, 8));
      const float mnew = fmaxf(mrun[r], mx);
      const float al   = __expf(mrun[r] - mnew);
      const float p0   = __expf(x0 - mnew);
      const float p1   = __expf(x1 - mnew);
      float ls = p0 + p1;
      ls += __shfl_xor(ls, 1);
      ls += __shfl_xor(ls, 2);
      ls += __shfl_xor(ls, 4);
      ls += __shfl_xor(ls, 8);
      mrun[r]  = mnew;
      lrun[r]  = lrun[r] * al + ls;
      alpha[r] = al;
      plds[(hb * 8 + r) * 32 + lr]      = (_Float16)p0;
      plds[(hb * 8 + r) * 32 + 16 + lr] = (_Float16)p1;
    }

    // rescale O accumulators
    #pragma unroll
    for (int dt = 0; dt < 4; ++dt)
      #pragma unroll
      for (int r = 0; r < 8; ++r)
        o[dt][r] *= alpha[r];

    __syncthreads();   // single-wave WG: cheap dscnt fence
    const v16h ap = cat16(*(const v8h*)&plds[lr * 32 + hb * 16],
                          *(const v8h*)&plds[lr * 32 + hb * 16 + 8]);
    __syncthreads();

    // ---- O += P V from the LDS V^T tile
    #pragma unroll
    for (int dt = 0; dt < 4; ++dt) {
      const _Float16* vrow = &vbuf[cur][(dt * 16 + lr) * 32 + hb * 16];
      o[dt] = wmma_f16(ap, *(const v16h*)vrow, o[dt]);
    }
  }

  // ---- epilogue: normalize, scatter to [b, s, h*DH + d] (f32)
  #pragma unroll
  for (int r = 0; r < 8; ++r) {
    const float inv = 1.0f / lrun[r];
    const int   m   = hb * 8 + r;
    float* orow = out + ((size_t)b * SEQ + q0 + m) * NOUT + h * DH;
    #pragma unroll
    for (int dt = 0; dt < 4; ++dt)
      orow[dt * 16 + lr] = o[dt][r] * inv;
  }
}

// ---------------------------------------------------------------------------
extern "C" void kernel_launch(void* const* d_in, const int* in_sizes, int n_in,
                              void* d_out, int out_size, void* d_ws, size_t ws_size,
                              hipStream_t stream) {
  (void)in_sizes; (void)n_in; (void)out_size; (void)ws_size;

  const float* x    = (const float*)d_in[0];
  const float* mask = (const float*)d_in[1];
  const float* Wq   = (const float*)d_in[2];
  const float* bq   = (const float*)d_in[3];
  const float* Wk   = (const float*)d_in[4];
  const float* bk   = (const float*)d_in[5];
  const float* Wv   = (const float*)d_in[6];
  const float* bv   = (const float*)d_in[7];
  float* out = (float*)d_out;

  // Workspace: Q, K (f16 [bh][s][d]), V^T (f16 [bh][d][s]) = 8MB each,
  // plus 3x transposed f16 weights (2MB each) = 30MB total.
  const size_t elems = (size_t)BS * NH * SEQ * DH;   // 4,194,304
  _Float16* qw = (_Float16*)d_ws;
  _Float16* kw = qw + elems;
  _Float16* vt = kw + elems;
  _Float16* wt = vt + elems;                          // 3 * 1M f16

  dim3 gT(DIN / 32, NOUT / 32, 3);
  wt_transpose_kernel<<<gT, dim3(32, 8), 0, stream>>>(Wq, Wk, Wv, wt);

  dim3 g1((BS * SEQ) / 16, NOUT / 64, 3);   // 256 x 16 x 3 strips
  qkv_proj_kernel<<<g1, dim3(32), 0, stream>>>(x, wt, bq, bk, bv, qw, kw, vt);

  dim3 g2(SEQ / 16, BS * NH);               // 128 x 32 query tiles
  flash_attn_kernel<<<g2, dim3(32), 0, stream>>>(mask, qw, kw, vt, out);
}
